// AutoformerEncoderLayer_36661840839445
// MI455X (gfx1250) — matrix operand order repaired and checked by reading, not compiled, not run on hardware
//
#include <hip/hip_runtime.h>

// ---------------------------------------------------------------------------
// Autoformer encoder layer for MI455X (gfx1250, wave32).
//   B=32, L=512, D=512, H=8, d=64, k=12, avgpool kernel=25
// FFN GEMMs: TDM (tensor_load_to_lds) stages the 64x512 f16 weight slice into
// LDS once per workgroup; 8 waves each compute a 32x64 tile with
// v_wmma_f32_16x16x32_f16 (A from global b128, B from LDS ds_load_b128).
// Autocorrelation is direct O(L^2) circular correlation from LDS (VALU).
// ---------------------------------------------------------------------------

typedef _Float16 half_t;
typedef __attribute__((ext_vector_type(16))) _Float16 v16h;
typedef __attribute__((ext_vector_type(8)))  _Float16 v8h;
typedef __attribute__((ext_vector_type(8)))  float    v8f;
typedef __attribute__((ext_vector_type(4)))  unsigned int v4u;
typedef __attribute__((ext_vector_type(4)))  int      v4i;
typedef __attribute__((ext_vector_type(8)))  int      v8i;

#define BB 32
#define LL 512
#define DD 512
#define HH 8
#define DH 64
#define KTOP 12

// -------------------------------------------------------------- f32 -> f16
__global__ __launch_bounds__(256) void cvt_f32_to_f16(const float* __restrict__ in,
                                                      half_t* __restrict__ out, int n) {
    int i = blockIdx.x * 256 + threadIdx.x;
    if (i < n) out[i] = (half_t)in[i];
}

// ------------------------------------------------- circular autocorrelation
// One block per (b, h, channel-group of 16). LDS tile sv[t][c], stride 17
// (gcd(17,64)=1 -> conflict-free column walks). Each thread owns one channel
// c and 32 lags. The rotated index u=(t+tau0)%512 is tracked incrementally so
// the common (no-wrap) path uses constant ds offsets p[j*17].
__global__ __launch_bounds__(256) void autocorr_kernel(const float* __restrict__ x,
                                                       float* __restrict__ corr) {
    __shared__ float sv[LL * 17];
    const int bid = blockIdx.x;          // 0 .. 1023
    const int cg  = bid & 3;             // channel group (16 channels)
    const int h   = (bid >> 2) & 7;
    const int b   = bid >> 5;
    const int dbase = h * DH + cg * 16;  // D-index of channel 0 in this group

    for (int i = 0; i < 32; ++i) {
        int idx = threadIdx.x + i * 256;     // 8192 elements
        int t = idx >> 4, c = idx & 15;
        sv[t * 17 + c] = x[((b * LL) + t) * DD + dbase + c];
    }
    __syncthreads();

    const int c    = threadIdx.x & 15;
    const int tau0 = (threadIdx.x >> 4) * 32;

    float acc[32];
#pragma unroll
    for (int j = 0; j < 32; ++j) acc[j] = 0.f;

    int u = tau0;                              // (t + tau0) mod 512
    for (int t = 0; t < LL; ++t) {
        float a = sv[t * 17 + c];
        if (u < LL - 31) {                     // no wrap inside 32-lag window
            const float* p = sv + u * 17 + c;
#pragma unroll
            for (int j = 0; j < 32; ++j) acc[j] += a * p[j * 17];
        } else {
#pragma unroll
            for (int j = 0; j < 32; ++j)
                acc[j] += a * sv[((u + j) & (LL - 1)) * 17 + c];
        }
        if (++u == LL) u = 0;
    }

    const int ch = b * DD + dbase + c;   // (b*8+h)*64 + dd  ==  b*512 + (h*64+dd)
#pragma unroll
    for (int j = 0; j < 32; ++j)
        corr[ch * LL + tau0 + j] = acc[j];
}

// ------------------------------------------------------- top-12 + softmax
__global__ __launch_bounds__(256) void topk_softmax_kernel(const float* __restrict__ corr,
                                                           float* __restrict__ wtop,
                                                           int* __restrict__ itop) {
    int ch = blockIdx.x * 256 + threadIdx.x;
    if (ch >= BB * DD) return;
    const float* row = corr + ch * LL;

    float vals[KTOP];
    int   idxs[KTOP];
#pragma unroll
    for (int i = 0; i < KTOP; ++i) { vals[i] = -3.4e38f; idxs[i] = 0; }

    for (int tau = 0; tau < LL; ++tau) {
        float v = row[tau];
        if (v > vals[KTOP - 1]) {          // strict '>' keeps stable tie order
            vals[KTOP - 1] = v; idxs[KTOP - 1] = tau;
#pragma unroll
            for (int i = KTOP - 1; i > 0; --i) {
                if (vals[i] > vals[i - 1]) {
                    float tv = vals[i]; vals[i] = vals[i - 1]; vals[i - 1] = tv;
                    int   ti = idxs[i]; idxs[i] = idxs[i - 1]; idxs[i - 1] = ti;
                }
            }
        }
    }

    float m = vals[0], e[KTOP], s = 0.f;
#pragma unroll
    for (int i = 0; i < KTOP; ++i) { e[i] = __expf(vals[i] - m); s += e[i]; }
    float inv = 1.f / s;
#pragma unroll
    for (int i = 0; i < KTOP; ++i) {
        wtop[ch * KTOP + i] = e[i] * inv;
        itop[ch * KTOP + i] = idxs[i];
    }
}

// --------------------------------- delay aggregation + attention residual
__global__ __launch_bounds__(256) void aggregate_kernel(const float* __restrict__ x,
                                                        const float* __restrict__ wtop,
                                                        const int* __restrict__ itop,
                                                        float* __restrict__ y) {
    const int bt = blockIdx.x;           // b*512 + t
    const int b  = bt >> 9;
    const int t  = bt & 511;
    for (int hd = threadIdx.x; hd < DD; hd += 256) {
        const int ch = b * DD + hd;
        float r = 0.f;
#pragma unroll
        for (int i = 0; i < KTOP; ++i) {
            int delay = itop[ch * KTOP + i];
            float w   = wtop[ch * KTOP + i];
            int gi = t + delay; if (gi > LL - 1) gi = LL - 1;
            r += w * x[(b * LL + gi) * DD + hd];
        }
        y[bt * DD + hd] = r + x[bt * DD + hd];
    }
}

// ----------------------- series decomposition: out = z - movavg25(z)
__global__ __launch_bounds__(256) void decomp_kernel(const float* __restrict__ z,
                                                     float* __restrict__ out_f,
                                                     half_t* __restrict__ out_h) {
    int e = blockIdx.x * 256 + threadIdx.x;          // over B*L*D
    int dch = e & 511;
    int l   = (e >> 9) & 511;
    const float* col = z + ((e >> 18) << 18) + dch;  // z[b, :, dch], stride 512
    int lo = l - 12; if (lo < 0) lo = 0;
    int hi = l + 12; if (hi > LL - 1) hi = LL - 1;
    float s = 0.f;
    for (int j = lo; j <= hi; ++j) s += col[j * DD];
    float v = z[e] - s * (1.0f / 25.0f);
    if (out_f) out_f[e] = v;
    if (out_h) out_h[e] = (half_t)v;
}

// -------------------------------------------------------- WMMA GEMM (FFN)
// C[m,n] = sum_k A[m,k] * W[n,k]  (+bias[n]) (+relu) (+resid[m,n])
// Workgroup: 8 waves, macro-tile 256(m) x 64(n). TDM loads W[n0:n0+64][0:512]
// (64 KB f16) into LDS once; each wave computes a 32x64 tile: 2 A-frags x
// 4 B-frags -> 8 WMMAs per 32-k chunk.
__global__ __launch_bounds__(256) void gemm_wmma_kernel(const half_t* __restrict__ A,
                                                        const half_t* __restrict__ W,
                                                        const float* __restrict__ bias,
                                                        const float* __restrict__ resid,
                                                        half_t* __restrict__ out_h,
                                                        float* __restrict__ out_f,
                                                        int do_relu) {
    const int K = DD, N = DD;
    __shared__ half_t bsm[64 * DD];              // 64 KB: W rows n0..n0+63

    const int blkN = blockIdx.x & 7;             // 8 n-blocks of 64
    const int blkM = blockIdx.x >> 3;            // 64 m-blocks of 256
    const int n0   = blkN * 64;
    const int wv   = threadIdx.x >> 5;           // wave id 0..7
    const int lane = threadIdx.x & 31;
    const int nl   = lane & 15;
    const int g    = lane >> 4;
    const int m0   = blkM * 256 + wv * 32;       // this wave's 32 rows

    // ---- TDM: stage weight slice into LDS (issued by wave 0 only) ----
    if (wv == 0) {
        unsigned long long ga = (unsigned long long)(const void*)(W + (unsigned)n0 * K);
        unsigned int lds = (unsigned int)(unsigned long long)(const void*)&bsm[0];
        // D# group0 (ISA 8.3): count=1 | lds_addr | global_addr(57b) | type=2
        v4u g0 = { 1u, lds, (unsigned int)ga,
                   (unsigned int)((ga >> 32) & 0x1FFFFFFu) | (2u << 30) };
        // D# group1 (ISA 8.4): data_size=2B; tensor 512x512 stride 512;
        // tile_dim0=512 (k, contiguous), tile_dim1=64 (n rows)
        v8i g1 = { (int)0x10000u,            // data_size=1 (2 bytes)
                   (int)(512u << 16),        // tensor_dim0 = 512 (lo16 here)
                   (int)(512u << 16),        // dim0 hi=0 | tensor_dim1=512 lo16
                   (int)(512u << 16),        // dim1 hi=0 | tile_dim0=512
                   64,                       // tile_dim1=64 | tile_dim2=0
                   512,                      // tensor_dim0_stride lo32
                   0, 0 };
        v4i gz = { 0, 0, 0, 0 };
#if __has_include(<hip/amd_detail/amd_gfx1250_TDM.h>)
        v8i gz8 = { 0, 0, 0, 0, 0, 0, 0, 0 };
        __builtin_amdgcn_tensor_load_to_lds(g0, g1, gz, gz, gz8, 0);
#else
        __builtin_amdgcn_tensor_load_to_lds(g0, g1, gz, gz, 0);
#endif
        __builtin_amdgcn_s_wait_tensorcnt(0);
    }
    __syncthreads();

    const half_t* __restrict__ arow0 = A + (m0 + nl) * K;        // A-frag ai=0
    const half_t* __restrict__ arow1 = A + (m0 + 16 + nl) * K;   // A-frag ai=1

    union Frag { v16h v; v8h h[2]; };
    v8f acc[2][4];
#pragma unroll
    for (int ai = 0; ai < 2; ++ai)
#pragma unroll
        for (int bi = 0; bi < 4; ++bi) acc[ai][bi] = (v8f){};

    for (int k0 = 0; k0 < K; k0 += 32) {
        Frag a0, a1;
        // A 16x32 f16 layout: elems 0..7 -> k0+8g+e ; elems 8..15 -> k0+16+8g+e'
        a0.h[0] = *(const v8h*)(arow0 + k0 + 8 * g);
        a0.h[1] = *(const v8h*)(arow0 + k0 + 16 + 8 * g);
        a1.h[0] = *(const v8h*)(arow1 + k0 + 8 * g);
        a1.h[1] = *(const v8h*)(arow1 + k0 + 16 + 8 * g);
#pragma unroll
        for (int bi = 0; bi < 4; ++bi) {
            // B 32x16 f16 layout: elems 0..7 -> k0+8g+e ; elems 8..15 -> k0+16+8g+e'
            Frag bfr;
            const half_t* p = bsm + (bi * 16 + nl) * K + k0 + 8 * g;
            bfr.h[0] = *(const v8h*)(p);
            bfr.h[1] = *(const v8h*)(p + 16);
            acc[0][bi] = __builtin_amdgcn_wmma_f32_16x16x32_f16(
                false, a0.v, false, bfr.v, (short)0, acc[0][bi], false, false);
            acc[1][bi] = __builtin_amdgcn_wmma_f32_16x16x32_f16(
                false, a1.v, false, bfr.v, (short)0, acc[1][bi], false, false);
        }
    }

#pragma unroll
    for (int ai = 0; ai < 2; ++ai)
#pragma unroll
        for (int bi = 0; bi < 4; ++bi) {
            int n = n0 + bi * 16 + nl;
            float bn = bias[n];
#pragma unroll
            for (int r = 0; r < 8; ++r) {        // D elem r -> m = r + 8g
                int m = m0 + ai * 16 + r + 8 * g;
                float v = acc[ai][bi][r] + bn;
                if (do_relu) v = fmaxf(v, 0.f);
                if (resid)  v += resid[m * N + n];
                if (out_h)  out_h[m * N + n] = (half_t)v;
                if (out_f)  out_f[m * N + n] = v;
            }
        }
}

// ---------------------------------------------------------------------------
extern "C" void kernel_launch(void* const* d_in, const int* in_sizes, int n_in,
                              void* d_out, int out_size, void* d_ws, size_t ws_size,
                              hipStream_t stream) {
    const float* x  = (const float*)d_in[0];   // [32,512,512]
    const float* w1 = (const float*)d_in[1];   // [512,512]
    const float* b1 = (const float*)d_in[2];   // [512]
    const float* w2 = (const float*)d_in[3];   // [512,512]
    const float* b2 = (const float*)d_in[4];   // [512]
    float* out = (float*)d_out;                // [32,512,512]

    char* ws = (char*)d_ws;
    float*  corr = (float*)(ws + 0);               //  32 MB corr[16384][512]
    float*  ybuf = (float*)(ws + 33554432u);       //  32 MB y (reused as z)
    float*  zbuf = ybuf;                           //  (ybuf dead before gemm2)
    float*  xs_f = (float*)(ws + 67108864u);       //  32 MB seasonal part f32
    half_t* xs_h = (half_t*)(ws + 100663296u);     //  16 MB seasonal part f16
    half_t* h_h  = (half_t*)(ws + 117440512u);     //  16 MB relu(xs@W1^T+b1) f16
    float*  wtop = (float*)(ws + 134217728u);      //  768 KB softmax weights
    int*    itop = (int*)  (ws + 135004160u);      //  768 KB delays
    half_t* w1h  = (half_t*)(ws + 135790592u);     //  512 KB W1 f16
    half_t* w2h  = (half_t*)(ws + 136314880u);     //  512 KB W2 f16

    const int NW = DD * DD;                        // 262144 weights per matrix

    cvt_f32_to_f16<<<(NW + 255) / 256, 256, 0, stream>>>(w1, w1h, NW);
    cvt_f32_to_f16<<<(NW + 255) / 256, 256, 0, stream>>>(w2, w2h, NW);

    // attention: circular autocorrelation -> top-12 softmax -> delay aggregate
    autocorr_kernel<<<BB * HH * 4, 256, 0, stream>>>(x, corr);
    topk_softmax_kernel<<<(BB * DD + 255) / 256, 256, 0, stream>>>(corr, wtop, itop);
    aggregate_kernel<<<BB * LL, 256, 0, stream>>>(x, wtop, itop, ybuf);

    // xs = y - movavg(y)   (f32 for residual, f16 for GEMM A)
    decomp_kernel<<<(BB * LL * DD) / 256, 256, 0, stream>>>(ybuf, xs_f, xs_h);

    // FFN: 512 workgroups, macro-tile 256x64, TDM-staged weights in LDS
    const int gemm_blocks = (16384 / 256) * (DD / 64);   // 64 * 8 = 512
    gemm_wmma_kernel<<<gemm_blocks, 256, 0, stream>>>(xs_h, w1h, b1, nullptr,
                                                      h_h, nullptr, 1);
    gemm_wmma_kernel<<<gemm_blocks, 256, 0, stream>>>(h_h, w2h, b2, xs_f,
                                                      nullptr, zbuf, 0);

    // out = z - movavg(z)
    decomp_kernel<<<(BB * LL * DD) / 256, 256, 0, stream>>>(zbuf, out, nullptr);
}
